// VisionBlock_60447369724195
// MI455X (gfx1250) — compile-verified, matrix operand order checked by reading, not run on hardware
//
#include <hip/hip_runtime.h>

typedef __attribute__((ext_vector_type(16))) __bf16 v16bf;
typedef __attribute__((ext_vector_type(8)))  __bf16 v8bf;
typedef __attribute__((ext_vector_type(8)))  float  v8f;

union FragBf { v16bf v; v8bf h[2]; };

// ---------------- problem constants ----------------
constexpr int S   = 16384;
constexpr int H   = 1152;
constexpr int NH  = 16;
constexpr int HD  = 72;
constexpr int WIN = 64;
constexpr int Iin = 4304;
constexpr int IP  = 4352;   // padded intermediate (divisible by 128)
constexpr int H3  = 3 * H;  // 3456
constexpr float EPS = 1e-6f;
constexpr float ATT_SCALE = 0.1178511301977579f; // 72^-0.5

// ---------------- async global->LDS staging (CDNA5) ----------------
// GLOBAL_LOAD_ASYNC_TO_LDS_B128 (ISA 15.18.3 op 98), GV mode: per-lane
// 16B copy MEM[vaddr] -> LDS[vdst]. Tracked by ASYNCcnt. Generic LDS
// pointers hold the LDS byte offset in their low 32 bits (ISA 10.2).
#define USE_ASYNC 1

__device__ __forceinline__ void copy16_g2l(const void* g, void* l) {
#if USE_ASYNC
    unsigned lds = (unsigned)(unsigned long long)l;
    asm volatile("global_load_async_to_lds_b128 %0, %1, off"
                 :: "v"(lds), "v"(g) : "memory");
#else
    *(v8bf*)l = *(const v8bf*)g;
#endif
}
__device__ __forceinline__ void async_drain() {
#if USE_ASYNC
    asm volatile("s_wait_asynccnt 0x0" ::: "memory");
#endif
}

// ---------------- weight transpose+pad fp32 -> bf16 ----------------
// dst is NP x KP row-major (dst[n*KP + k] = src[k*N_ + n]), zero padded.
__global__ void cvt_transpose_kernel(const float* __restrict__ src, __bf16* __restrict__ dst,
                                     int K_, int N_, int KP, int NP) {
    int i = blockIdx.x * 256 + threadIdx.x;
    if (i >= NP * KP) return;
    int n = i / KP, k = i % KP;
    dst[i] = (__bf16)((n < N_ && k < K_) ? src[(size_t)k * N_ + n] : 0.0f);
}

__global__ void pad_bias_kernel(const float* __restrict__ src, float* __restrict__ dst,
                                int N_, int NP) {
    int i = blockIdx.x * 256 + threadIdx.x;
    if (i < NP) dst[i] = (i < N_) ? src[i] : 0.0f;
}

// ---------------- RMSNorm (fp32 in -> bf16 out) ----------------
__global__ __launch_bounds__(256)
void rmsnorm_kernel(const float* __restrict__ x, const float* __restrict__ g,
                    __bf16* __restrict__ out) {
    __shared__ float red[256];
    int row = blockIdx.x;
    const float* xr = x + (size_t)row * H;
    float s = 0.0f;
    for (int i = threadIdx.x; i < H; i += 256) { float v = xr[i]; s += v * v; }
    red[threadIdx.x] = s;
    __syncthreads();
    for (int off = 128; off > 0; off >>= 1) {
        if ((int)threadIdx.x < off) red[threadIdx.x] += red[threadIdx.x + off];
        __syncthreads();
    }
    float rs = rsqrtf(red[0] / (float)H + EPS);
    __bf16* orow = out + (size_t)row * H;
    for (int i = threadIdx.x; i < H; i += 256) orow[i] = (__bf16)(xr[i] * g[i] * rs);
}

// ---------------- RoPE in place on q,k halves of qkv (bf16) ----------------
__global__ void rope_kernel(__bf16* __restrict__ qkv, const float* __restrict__ cosb,
                            const float* __restrict__ sinb) {
    int i = blockIdx.x * blockDim.x + threadIdx.x;
    const int total = S * NH * (HD / 2);
    if (i >= total) return;
    int d = i % (HD / 2);
    int tmp = i / (HD / 2);
    int hh = tmp % NH;
    int sidx = tmp / NH;
    float c1 = cosb[sidx * HD + d],      s1 = sinb[sidx * HD + d];
    float c2 = cosb[sidx * HD + d + 36], s2 = sinb[sidx * HD + d + 36];
    size_t base = (size_t)sidx * H3 + hh * HD;
    float q1 = (float)qkv[base + d], q2 = (float)qkv[base + d + 36];
    qkv[base + d]      = (__bf16)(q1 * c1 - q2 * s1);
    qkv[base + d + 36] = (__bf16)(q2 * c2 + q1 * s2);
    size_t kb = base + H;
    float k1 = (float)qkv[kb + d], k2 = (float)qkv[kb + d + 36];
    qkv[kb + d]      = (__bf16)(k1 * c1 - k2 * s1);
    qkv[kb + d + 36] = (__bf16)(k2 * c2 + k1 * s2);
}

// ---------------- bf16 WMMA GEMM, 128x128x64 tiles, double buffered ----------
// A: M x K row-major. Bt: N x K row-major (pre-transposed weight).
// MODE 0: outb = bf16(acc + bias)
// MODE 1: outf = acc + bias + res          (fp32 residual fusion)
// MODE 2: outb = bf16(silu(acc+bias) * outb)   (gate*up fusion, in place)
constexpr int TSTR = 72;   // LDS row stride for 64-K tiles (16B aligned, bank-spread)

__device__ __forceinline__ void stage_tile(const __bf16* __restrict__ src, int ld,
                                           int rowBase, int kb, __bf16* buf, int t) {
#pragma unroll
    for (int c = 0; c < 4; ++c) {
        int lin = (c * 256 + t) * 8;
        int row = lin >> 6;
        int kk  = lin & 63;
        copy16_g2l(src + (size_t)(rowBase + row) * ld + kb + kk, buf + row * TSTR + kk);
    }
}

template <int MODE>
__global__ __launch_bounds__(256)
void gemm_bf16_kernel(const __bf16* __restrict__ A, const __bf16* __restrict__ Bt,
                      const float* __restrict__ bias, const float* __restrict__ res,
                      __bf16* __restrict__ outb, float* __restrict__ outf,
                      int M, int N, int K) {
    __shared__ __bf16 As[2][128 * TSTR];
    __shared__ __bf16 Bs[2][128 * TSTR];
    const int t    = threadIdx.x;
    const int lane = t & 31;
    const int wave = t >> 5;
    const int hlf  = lane >> 4;
    const int r    = lane & 15;
    const int wm   = wave >> 2;       // 0..1 -> 64 rows each
    const int wn   = wave & 3;        // 0..3 -> 32 cols each
    const int mBase = blockIdx.y * 128;
    const int nBase = blockIdx.x * 128;

    v8f acc[4][2] = {};

    stage_tile(A,  K, mBase, 0, As[0], t);
    stage_tile(Bt, K, nBase, 0, Bs[0], t);
    async_drain();
    __syncthreads();

    int cur = 0;
    for (int kb = 0; kb < K; kb += 64) {
        int nxt = cur ^ 1;
        if (kb + 64 < K) {
            stage_tile(A,  K, mBase, kb + 64, As[nxt], t);
            stage_tile(Bt, K, nBase, kb + 64, Bs[nxt], t);
        }
        const __bf16* a = As[cur];
        const __bf16* b = Bs[cur];
#pragma unroll
        for (int ks = 0; ks < 2; ++ks) {
            FragBf afrag[4];
#pragma unroll
            for (int mt = 0; mt < 4; ++mt) {
                int row = wm * 64 + mt * 16 + r;
                afrag[mt].h[0] = *(const v8bf*)(&a[row * TSTR + ks * 32 + hlf * 8]);
                afrag[mt].h[1] = *(const v8bf*)(&a[row * TSTR + ks * 32 + 16 + hlf * 8]);
            }
#pragma unroll
            for (int nt = 0; nt < 2; ++nt) {
                int col = wn * 32 + nt * 16 + r;
                FragBf bfrag;
                bfrag.h[0] = *(const v8bf*)(&b[col * TSTR + ks * 32 + hlf * 8]);
                bfrag.h[1] = *(const v8bf*)(&b[col * TSTR + ks * 32 + 16 + hlf * 8]);
#pragma unroll
                for (int mt = 0; mt < 4; ++mt) {
                    acc[mt][nt] = __builtin_amdgcn_wmma_f32_16x16x32_bf16(
                        false, afrag[mt].v, false, bfrag.v, (short)0, acc[mt][nt], false, false);
                }
            }
        }
        async_drain();
        __syncthreads();
        cur = nxt;
    }

    // epilogue: C layout — vgpr j: lanes 0-15 row j, lanes 16-31 row j+8; lane r = col
#pragma unroll
    for (int mt = 0; mt < 4; ++mt) {
#pragma unroll
        for (int nt = 0; nt < 2; ++nt) {
            int col = nBase + wn * 32 + nt * 16 + r;
            float bv = bias[col];
#pragma unroll
            for (int j = 0; j < 8; ++j) {
                int row = mBase + wm * 64 + mt * 16 + j + hlf * 8;
                size_t idx = (size_t)row * N + col;
                float v = acc[mt][nt][j] + bv;
                if (MODE == 0) {
                    outb[idx] = (__bf16)v;
                } else if (MODE == 1) {
                    outf[idx] = v + res[idx];
                } else {
                    float u = (float)outb[idx];
                    float sg = v / (1.0f + __expf(-v));
                    outb[idx] = (__bf16)(sg * u);
                }
            }
        }
    }
}

// ---------------- windowed attention: one WG per (window, head) ----------------
constexpr int KQS = 104;  // Q/K LDS stride (96 valid K, padded, 16B-aligned rows)
constexpr int PSS = 72;   // P stride (64 valid K)
constexpr int VTS = 72;   // V^T stride (64 valid tokens)

__global__ __launch_bounds__(128)
void attn_kernel(const __bf16* __restrict__ qkv, __bf16* __restrict__ o) {
    __shared__ __bf16 Qs[64 * KQS];
    __shared__ __bf16 Ks[64 * KQS];
    __shared__ __bf16 Ps[64 * PSS];
    __shared__ __bf16 Vt[80 * VTS];
    const int t  = threadIdx.x;
    const int w  = blockIdx.x / NH;
    const int hh = blockIdx.x % NH;
    const int rowBase = w * WIN;

    // zero the K-dim pads (dims 72..95 of Q/K, rows 72..79 of V^T)
    for (int i = t; i < 64 * 24; i += 128) {
        int token = i / 24, d = 72 + (i % 24);
        Qs[token * KQS + d] = (__bf16)0.0f;
        Ks[token * KQS + d] = (__bf16)0.0f;
    }
    for (int i = t; i < 8 * VTS; i += 128)
        Vt[(72 + i / VTS) * VTS + (i % VTS)] = (__bf16)0.0f;

    // load Q,K row-major; V transposed (Vt[dim][token])
    for (int c = t; c < 64 * 9; c += 128) {
        int token = c / 9, d8 = (c % 9) * 8;
        size_t g = (size_t)(rowBase + token) * H3 + hh * HD + d8;
        *(v8bf*)(&Qs[token * KQS + d8]) = *(const v8bf*)(qkv + g);
        *(v8bf*)(&Ks[token * KQS + d8]) = *(const v8bf*)(qkv + g + H);
        v8bf vv = *(const v8bf*)(qkv + g + 2 * H);
#pragma unroll
        for (int j = 0; j < 8; ++j) Vt[(d8 + j) * VTS + token] = vv[j];
    }
    __syncthreads();

    const int lane = t & 31, hlf = lane >> 4, r = lane & 15;
    const int wave = t >> 5;        // m-tile (16 query rows per wave)
    const int mrow = wave * 16;

    // S = Q K^T  (M=64 split across waves, N=64, K=96)
    v8f sacc[4] = {};
    for (int ks = 0; ks < 3; ++ks) {
        FragBf a;
        a.h[0] = *(const v8bf*)(&Qs[(mrow + r) * KQS + ks * 32 + hlf * 8]);
        a.h[1] = *(const v8bf*)(&Qs[(mrow + r) * KQS + ks * 32 + 16 + hlf * 8]);
#pragma unroll
        for (int nt = 0; nt < 4; ++nt) {
            FragBf b;
            b.h[0] = *(const v8bf*)(&Ks[(nt * 16 + r) * KQS + ks * 32 + hlf * 8]);
            b.h[1] = *(const v8bf*)(&Ks[(nt * 16 + r) * KQS + ks * 32 + 16 + hlf * 8]);
            sacc[nt] = __builtin_amdgcn_wmma_f32_16x16x32_bf16(
                false, a.v, false, b.v, (short)0, sacc[nt], false, false);
        }
    }

    // softmax over the 64 columns of each row (16 lanes/half x 4 tiles)
#pragma unroll
    for (int j = 0; j < 8; ++j) {
        float mx = -1e30f;
#pragma unroll
        for (int nt = 0; nt < 4; ++nt) {
            float v = sacc[nt][j] * ATT_SCALE; sacc[nt][j] = v; mx = fmaxf(mx, v);
        }
        for (int m = 1; m <= 8; m <<= 1) mx = fmaxf(mx, __shfl_xor(mx, m, 32));
        float sum = 0.0f;
#pragma unroll
        for (int nt = 0; nt < 4; ++nt) {
            float e = __expf(sacc[nt][j] - mx); sacc[nt][j] = e; sum += e;
        }
        for (int m = 1; m <= 8; m <<= 1) sum += __shfl_xor(sum, m, 32);
        float inv = 1.0f / sum;
        int prow = mrow + j + hlf * 8;
#pragma unroll
        for (int nt = 0; nt < 4; ++nt)
            Ps[prow * PSS + nt * 16 + r] = (__bf16)(sacc[nt][j] * inv);
    }
    __syncthreads();

    // O = P V  (M=16 per wave, N=80 padded, K=64)
    v8f oacc[5] = {};
    for (int ks = 0; ks < 2; ++ks) {
        FragBf a;
        a.h[0] = *(const v8bf*)(&Ps[(mrow + r) * PSS + ks * 32 + hlf * 8]);
        a.h[1] = *(const v8bf*)(&Ps[(mrow + r) * PSS + ks * 32 + 16 + hlf * 8]);
#pragma unroll
        for (int nt = 0; nt < 5; ++nt) {
            FragBf b;
            b.h[0] = *(const v8bf*)(&Vt[(nt * 16 + r) * VTS + ks * 32 + hlf * 8]);
            b.h[1] = *(const v8bf*)(&Vt[(nt * 16 + r) * VTS + ks * 32 + 16 + hlf * 8]);
            oacc[nt] = __builtin_amdgcn_wmma_f32_16x16x32_bf16(
                false, a.v, false, b.v, (short)0, oacc[nt], false, false);
        }
    }
#pragma unroll
    for (int nt = 0; nt < 5; ++nt) {
        int col = nt * 16 + r;
        if (col < HD) {
#pragma unroll
            for (int j = 0; j < 8; ++j) {
                int row = rowBase + mrow + j + hlf * 8;
                o[(size_t)row * H + hh * HD + col] = (__bf16)oacc[nt][j];
            }
        }
    }
}

// ---------------- launcher ----------------
extern "C" void kernel_launch(void* const* d_in, const int* in_sizes, int n_in,
                              void* d_out, int out_size, void* d_ws, size_t ws_size,
                              hipStream_t stream) {
    (void)in_sizes; (void)n_in; (void)out_size; (void)ws_size;
    const float* x     = (const float*)d_in[0];
    const float* cosb  = (const float*)d_in[1];
    const float* sinb  = (const float*)d_in[2];
    const float* g1    = (const float*)d_in[3];
    const float* g2    = (const float*)d_in[4];
    const float* Wqkv  = (const float*)d_in[5];
    const float* bqkv  = (const float*)d_in[6];
    const float* Wproj = (const float*)d_in[7];
    const float* bproj = (const float*)d_in[8];
    const float* Wg    = (const float*)d_in[9];
    const float* bg    = (const float*)d_in[10];
    const float* Wu    = (const float*)d_in[11];
    const float* bu    = (const float*)d_in[12];
    const float* Wd    = (const float*)d_in[13];
    const float* bd    = (const float*)d_in[14];
    float* out = (float*)d_out;

    char* ws = (char*)d_ws;
    size_t off = 0;
    auto alloc = [&](size_t bytes) -> void* {
        void* p = ws + off;
        off += (bytes + 255) & ~(size_t)255;
        return p;
    };
    __bf16* h_bf    = (__bf16*)alloc((size_t)S * H * 2);
    __bf16* wqkv_t  = (__bf16*)alloc((size_t)H3 * H * 2);   // [3456][1152]
    __bf16* wproj_t = (__bf16*)alloc((size_t)H * H * 2);    // [1152][1152]
    __bf16* wg_t    = (__bf16*)alloc((size_t)IP * H * 2);   // [4352][1152]
    __bf16* wu_t    = (__bf16*)alloc((size_t)IP * H * 2);   // [4352][1152]
    __bf16* wd_t    = (__bf16*)alloc((size_t)H * IP * 2);   // [1152][4352]
    float*  bg_pad  = (float*)alloc((size_t)IP * 4);
    float*  bu_pad  = (float*)alloc((size_t)IP * 4);
    __bf16* qkv_bf  = (__bf16*)alloc((size_t)S * H3 * 2);
    __bf16* o_bf    = (__bf16*)alloc((size_t)S * H * 2);
    float*  x2      = (float*)alloc((size_t)S * H * 4);
    __bf16* h2_bf   = (__bf16*)alloc((size_t)S * H * 2);
    __bf16* act_bf  = (__bf16*)alloc((size_t)S * IP * 2);

    auto blocks = [](long n) { return (int)((n + 255) / 256); };

    // weight transpose + pad (fp32 -> bf16, [N][K] layout)
    cvt_transpose_kernel<<<blocks((long)H3 * H), 256, 0, stream>>>(Wqkv, wqkv_t, H, H3, H, H3);
    cvt_transpose_kernel<<<blocks((long)H * H), 256, 0, stream>>>(Wproj, wproj_t, H, H, H, H);
    cvt_transpose_kernel<<<blocks((long)IP * H), 256, 0, stream>>>(Wg, wg_t, H, Iin, H, IP);
    cvt_transpose_kernel<<<blocks((long)IP * H), 256, 0, stream>>>(Wu, wu_t, H, Iin, H, IP);
    cvt_transpose_kernel<<<blocks((long)H * IP), 256, 0, stream>>>(Wd, wd_t, Iin, H, IP, H);
    pad_bias_kernel<<<blocks(IP), 256, 0, stream>>>(bg, bg_pad, Iin, IP);
    pad_bias_kernel<<<blocks(IP), 256, 0, stream>>>(bu, bu_pad, Iin, IP);

    // 1) h = rmsnorm(x, g1)
    rmsnorm_kernel<<<S, 256, 0, stream>>>(x, g1, h_bf);

    // 2) qkv = h @ Wqkv + bqkv   (bf16 out)
    gemm_bf16_kernel<0><<<dim3(H3 / 128, S / 128), 256, 0, stream>>>(
        h_bf, wqkv_t, bqkv, nullptr, qkv_bf, nullptr, S, H3, H);

    // 3) RoPE on q,k
    rope_kernel<<<blocks((long)S * NH * (HD / 2)), 256, 0, stream>>>(qkv_bf, cosb, sinb);

    // 4) windowed attention -> o (bf16)
    attn_kernel<<<(S / WIN) * NH, 128, 0, stream>>>(qkv_bf, o_bf);

    // 5) x2 = x + o @ Wproj + bproj   (fp32 out, residual fused)
    gemm_bf16_kernel<1><<<dim3(H / 128, S / 128), 256, 0, stream>>>(
        o_bf, wproj_t, bproj, x, nullptr, x2, S, H, H);

    // 6) h2 = rmsnorm(x2, g2)
    rmsnorm_kernel<<<S, 256, 0, stream>>>(x2, g2, h2_bf);

    // 7) act = h2 @ Wu + bu  (bf16)
    gemm_bf16_kernel<0><<<dim3(IP / 128, S / 128), 256, 0, stream>>>(
        h2_bf, wu_t, bu_pad, nullptr, act_bf, nullptr, S, IP, H);

    // 8) act = silu(h2 @ Wg + bg) * act   (fused epilogue)
    gemm_bf16_kernel<2><<<dim3(IP / 128, S / 128), 256, 0, stream>>>(
        h2_bf, wg_t, bg_pad, nullptr, act_bf, nullptr, S, IP, H);

    // 9) out = x2 + act @ Wd + bd   (fp32 out, residual fused)
    gemm_bf16_kernel<1><<<dim3(H / 128, S / 128), 256, 0, stream>>>(
        act_bf, wd_t, bd, x2, nullptr, out, S, H, IP);
}
